// DeformAtten1D_35356170780738
// MI455X (gfx1250) — matrix operand order repaired and checked by reading, not compiled, not run on hardware
//
#include <hip/hip_runtime.h>

typedef __attribute__((ext_vector_type(16))) _Float16 v16h;
typedef __attribute__((ext_vector_type(8)))  float    v8f;
typedef __attribute__((ext_vector_type(8)))  _Float16 h8;

constexpr int Bc = 4, Lc = 4096, Cc = 1024, Hc = 16, Gc = 4, Kc = 5;
constexpr int Cg = Cc / Gc;   // 256
constexpr int Dh = Cc / Hc;   // 64
constexpr int BG = Bc * Gc;   // 16
constexpr int BH = Bc * Hc;   // 64

#define LDS_ST 40  // halfs per LDS tile row: 32 data + 8 pad; row = 80B -> 16B aligned chunks

// ---------------------------------------------------------------------------
// Generic batched WMMA GEMM:  Out[z] = scale * A[z] x B[z] + bias (+ rel)
//   A is MxK  (transA: A[m,k] = Asrc[k*lda + m], else Asrc[m*lda + k])
//   B given as Bt (NxK row-major) unless transB: Bt[n,k] = Bsrc[k*ldb + n]
//   Per-z offsets: z1 = z / zdiv, z2 = z % zdiv; off = off1*z1 + off2*z2
// REQUIRES: M % 64 == 0 (per grid), N % 64 == 0, K % 32 == 0, all base
// pointers + leading dims 16B aligned (true for every call in this workload).
// Block: 128 threads = 4 waves; block tile 64(M) x 64(N); wave tile 16x64.
// ---------------------------------------------------------------------------
__global__ __launch_bounds__(128) void wmma_gemm_kernel(
    const _Float16* __restrict__ A, const _Float16* __restrict__ Bt,
    void* __restrict__ Out, const float* __restrict__ bias,
    const float* __restrict__ rel, int relL,
    int M, int N, int K, int lda, int ldb, int ldo,
    int transA, int transB, int outF32, float scale,
    long long offA1, long long offA2, long long offB1, long long offB2,
    long long offO1, long long offO2, int zdiv)
{
  __shared__ _Float16 As[64 * LDS_ST];
  __shared__ _Float16 Bs[64 * LDS_ST];

  const int tid  = threadIdx.x;
  const int lane = tid & 31;
  const int w    = tid >> 5;      // wave id 0..3
  const int hi   = lane >> 4;     // lane half (0: lanes 0-15, 1: lanes 16-31)
  const int lm   = lane & 15;
  const int m0   = blockIdx.x * 64;
  const int n0   = blockIdx.y * 64;
  const int z    = blockIdx.z;
  const int z1   = z / zdiv, z2 = z % zdiv;

  const _Float16* Ab = A  + offA1 * z1 + offA2 * z2;
  const _Float16* Bb = Bt + offB1 * z1 + offB2 * z2;
  const long long offO = offO1 * z1 + offO2 * z2;

  v8f acc[4];
#pragma unroll
  for (int j = 0; j < 4; ++j)
#pragma unroll
    for (int r = 0; r < 8; ++r) acc[j][r] = 0.0f;

  for (int k0 = 0; k0 < K; k0 += 32) {
    __syncthreads();
    // ---- Stage A tile [64 x 32] f16 into LDS (b128 global loads) ----
    if (!transA) {
      // 64 rows x 4 chunks of 8 halfs = 256 vec units; 2 per thread
#pragma unroll
      for (int t = tid; t < 256; t += 128) {
        int mm = t >> 2, kc = (t & 3) * 8;
        h8 v = *(const h8*)(Ab + (long long)(m0 + mm) * lda + (k0 + kc));
        *(h8*)&As[mm * LDS_ST + kc] = v;
      }
    } else {
      // contiguous axis is m: load 8 halfs along m, scatter into LDS column
#pragma unroll
      for (int t = tid; t < 256; t += 128) {
        int kk = t >> 3, mg = (t & 7) * 8;
        h8 v = *(const h8*)(Ab + (long long)(k0 + kk) * lda + (m0 + mg));
#pragma unroll
        for (int i = 0; i < 8; ++i) As[(mg + i) * LDS_ST + kk] = v[i];
      }
    }
    // ---- Stage Bt tile [64 x 32] f16 into LDS ----
    if (!transB) {
#pragma unroll
      for (int t = tid; t < 256; t += 128) {
        int nn = t >> 2, kc = (t & 3) * 8;
        h8 v = *(const h8*)(Bb + (long long)(n0 + nn) * ldb + (k0 + kc));
        *(h8*)&Bs[nn * LDS_ST + kc] = v;
      }
    } else {
#pragma unroll
      for (int t = tid; t < 256; t += 128) {
        int kk = t >> 3, ng = (t & 7) * 8;
        h8 v = *(const h8*)(Bb + (long long)(k0 + kk) * ldb + (n0 + ng));
#pragma unroll
        for (int i = 0; i < 8; ++i) Bs[(ng + i) * LDS_ST + kk] = v[i];
      }
    }
    // Prefetch next A tile (gfx1250 global_prefetch path)
    if (k0 + 32 < K && tid < 64) {
      const _Float16* pn = transA
          ? &Ab[(long long)(k0 + 32 + (tid >> 1)) * lda + m0]
          : &Ab[(long long)(m0 + tid) * lda + (k0 + 32)];
      __builtin_prefetch(pn, 0, 1);
    }
    __syncthreads();

    // Build A fragment: 16x32 f16, M = lane%16 (+ wave row offset)
    // K pair (uint) index: 4*hi + 8*(g>=4) + (g&3)
    union { v16h v; unsigned u[8]; } af;
    const unsigned* ar = (const unsigned*)&As[(w * 16 + lm) * LDS_ST];
#pragma unroll
    for (int g = 0; g < 8; ++g)
      af.u[g] = ar[4 * hi + 8 * (g >> 2) + (g & 3)];

#pragma unroll
    for (int jn = 0; jn < 4; ++jn) {
      // B fragment: Bt row n = jn*16 + lane%16, K pair index 4*hi + 8*(g>=4) + (g&3)
      union { v16h v; unsigned u[8]; } bf;
      const unsigned* br = (const unsigned*)&Bs[(jn * 16 + lm) * LDS_ST];
#pragma unroll
      for (int g = 0; g < 8; ++g) bf.u[g] = br[4 * hi + 8 * (g >> 2) + (g & 3)];
      acc[jn] = __builtin_amdgcn_wmma_f32_16x16x32_f16(
          false, af.v, false, bf.v, (short)0, acc[jn], false, false);
    }
  }

  // Epilogue: C/D layout — VGPR r: M = r + 8*hi, N = lane%16
#pragma unroll
  for (int jn = 0; jn < 4; ++jn) {
    int n = n0 + jn * 16 + lm;
    float bval = bias ? bias[n] : 0.0f;
#pragma unroll
    for (int r = 0; r < 8; ++r) {
      int m = m0 + w * 16 + 8 * hi + r;
      float v = acc[jn][r] * scale + bval;
      if (rel) v += rel[(long long)n * relL + (m % relL)];
      long long oi = offO + (long long)m * ldo + n;
      if (outF32) ((float*)Out)[oi] = v;
      else        ((_Float16*)Out)[oi] = (_Float16)v;
    }
  }
}

// ---------------------------------------------------------------------------
__global__ void cast_f32_f16_kernel(const float* __restrict__ in,
                                    _Float16* __restrict__ out, long long n) {
  long long i = (long long)blockIdx.x * blockDim.x + threadIdx.x;
  if (i < n) out[i] = (_Float16)in[i];
}

// Fuse conv2(1x1) into conv1: weff[k][c] = sum_d W2[d]*W1[d,c,k]; weff[Kc*Cg]=beff
__global__ void compute_weff_kernel(const float* __restrict__ W1,
                                    const float* __restrict__ b1,
                                    const float* __restrict__ W2,
                                    const float* __restrict__ b2,
                                    float* __restrict__ weff) {
  int c = threadIdx.x;  // 256
  for (int k = 0; k < Kc; ++k) {
    float s = 0.0f;
    for (int d = 0; d < Cg; ++d) s += W2[d] * W1[((long long)d * Cg + c) * Kc + k];
    weff[k * Cg + c] = s;
  }
  if (c == 0) {
    float s = b2[0];
    for (int d = 0; d < Cg; ++d) s += W2[d] * b1[d];
    weff[Kc * Cg] = s;
  }
}

// off[bg, n] for n in [0, L): fused conv -> tanh * Kc
// conv output index n maps to valid center for n >= 2 (n < L+2 always holds)
__global__ void compute_off_kernel(const _Float16* __restrict__ qh,
                                   const float* __restrict__ weff,
                                   const float* __restrict__ b2,
                                   float* __restrict__ off) {
  __shared__ float wsh[Kc * Cg];
  int tid = threadIdx.x;
  for (int t = tid; t < Kc * Cg; t += 256) wsh[t] = weff[t];
  __syncthreads();
  int bg = blockIdx.y;
  int n  = blockIdx.x * 256 + tid;
  int b = bg / Gc, g = bg % Gc;
  float val;
  if (n < 2) {
    val = b2[0];  // pure zero-padding region of conv2 input
  } else {
    float acc = weff[Kc * Cg];  // beff
    const _Float16* qbase = qh + (long long)b * Lc * Cc + g * Cg;
#pragma unroll
    for (int k = 0; k < Kc; ++k) {
      int l = n - 4 + k;
      if (l < 0 || l >= Lc) continue;
      const _Float16* ql = qbase + (long long)l * Cc;
      const float* wk = &wsh[k * Cg];
      for (int c = 0; c < Cg; c += 8) {
        h8 qv = *(const h8*)(ql + c);
#pragma unroll
        for (int i = 0; i < 8; ++i) acc += wk[c + i] * (float)qv[i];
      }
    }
    val = acc;
  }
  off[(long long)bg * Lc + n] = tanhf(val) * (float)Kc;
}

// Bilinear deformable gather: xs_t[b, n, g*Cg + c]  (only n < L survives)
// pos = (n + off) * L/(L+3) - 0.5  (grid_sample align_corners=False, zeros pad)
__global__ void gather_xs_kernel(const float* __restrict__ x,
                                 const float* __restrict__ off,
                                 _Float16* __restrict__ xs) {
  int c  = threadIdx.x;   // 0..Cg-1
  int n  = blockIdx.x;    // 0..L-1
  int bg = blockIdx.y;    // 0..BG-1
  int b = bg / Gc, g = bg % Gc;
  float o   = off[(long long)bg * Lc + n];
  float pos = ((float)n + o) * ((float)Lc / (float)(Lc + 3)) - 0.5f;
  float fl  = floorf(pos);
  int   i0  = (int)fl;
  float fr  = pos - fl;
  const float* xb = x + (long long)b * Lc * Cc + g * Cg + c;
  float v0 = (i0 >= 0     && i0 < Lc)     ? xb[(long long)i0 * Cc]       : 0.0f;
  float v1 = (i0 + 1 >= 0 && i0 + 1 < Lc) ? xb[(long long)(i0 + 1) * Cc] : 0.0f;
  xs[((long long)b * Lc + n) * Cc + g * Cg + c] = (_Float16)((1.0f - fr) * v0 + fr * v1);
}

// Row softmax over 64 logits -> f16 attn
__global__ void softmax64_kernel(const float* __restrict__ logits,
                                 _Float16* __restrict__ attn) {
  __shared__ float red[64];
  int j = threadIdx.x;                // 0..63
  long long row = blockIdx.x;         // bh*64 + i
  float v = logits[row * 64 + j];
  red[j] = v;
  __syncthreads();
  for (int s = 32; s > 0; s >>= 1) {
    if (j < s) red[j] = fmaxf(red[j], red[j + s]);
    __syncthreads();
  }
  float mx = red[0];
  __syncthreads();
  red[j] = __expf(v - mx);
  __syncthreads();
  for (int s = 32; s > 0; s >>= 1) {
    if (j < s) red[j] += red[j + s];
    __syncthreads();
  }
  float sum = red[0];
  attn[row * 64 + j] = (_Float16)(__expf(v - mx) / sum);
}

// ---------------------------------------------------------------------------
extern "C" void kernel_launch(void* const* d_in, const int* in_sizes, int n_in,
                              void* d_out, int out_size, void* d_ws, size_t ws_size,
                              hipStream_t stream) {
  const float* x   = (const float*)d_in[0];
  const float* Wq  = (const float*)d_in[1];
  const float* bq  = (const float*)d_in[2];
  const float* Wk  = (const float*)d_in[3];
  const float* bk  = (const float*)d_in[4];
  const float* Wv  = (const float*)d_in[5];
  const float* bv  = (const float*)d_in[6];
  const float* Wo  = (const float*)d_in[7];
  const float* bo  = (const float*)d_in[8];
  const float* W1  = (const float*)d_in[9];
  const float* b1  = (const float*)d_in[10];
  const float* W2  = (const float*)d_in[11];
  const float* b2  = (const float*)d_in[12];
  const float* rel = (const float*)d_in[13];

  char* wsp = (char*)d_ws;
  auto alloc = [&](size_t bytes) -> char* {
    char* p = wsp;
    wsp += (bytes + 255) & ~(size_t)255;
    return p;
  };
  const size_t NBLC = (size_t)Bc * Lc * Cc;   // 16,777,216
  _Float16* xh   = (_Float16*)alloc(NBLC * 2);
  _Float16* qh   = (_Float16*)alloc(NBLC * 2);
  _Float16* xsh  = (_Float16*)alloc(NBLC * 2);
  _Float16* khh  = (_Float16*)alloc(NBLC * 2);
  _Float16* vhh  = (_Float16*)alloc(NBLC * 2);
  _Float16* oh   = (_Float16*)alloc(NBLC * 2);
  _Float16* Wqh  = (_Float16*)alloc((size_t)Cc * Cc * 2);
  _Float16* Wkh  = (_Float16*)alloc((size_t)Cc * Cc * 2);
  _Float16* Wvh  = (_Float16*)alloc((size_t)Cc * Cc * 2);
  _Float16* Woh  = (_Float16*)alloc((size_t)Cc * Cc * 2);
  float*    offb = (float*)   alloc((size_t)BG * Lc * 4);
  float*    weff = (float*)   alloc((size_t)(Kc * Cg + 1) * 4);
  float*    lgts = (float*)   alloc((size_t)BH * Dh * Dh * 4);
  _Float16* attn = (_Float16*)alloc((size_t)BH * Dh * Dh * 2);

  // 1) f32 -> f16 casts
  long long n1 = (long long)NBLC;
  cast_f32_f16_kernel<<<(unsigned)((n1 + 255) / 256), 256, 0, stream>>>(x, xh, n1);
  long long nw = (long long)Cc * Cc;
  cast_f32_f16_kernel<<<(unsigned)((nw + 255) / 256), 256, 0, stream>>>(Wq, Wqh, nw);
  cast_f32_f16_kernel<<<(unsigned)((nw + 255) / 256), 256, 0, stream>>>(Wk, Wkh, nw);
  cast_f32_f16_kernel<<<(unsigned)((nw + 255) / 256), 256, 0, stream>>>(Wv, Wvh, nw);
  cast_f32_f16_kernel<<<(unsigned)((nw + 255) / 256), 256, 0, stream>>>(Wo, Woh, nw);

  dim3 gBig(Bc * Lc / 64, Cc / 64, 1);  // 256 x 16

  // 2) Q = x @ Wq^T + bq   (store [B,L,C] f16)
  wmma_gemm_kernel<<<gBig, 128, 0, stream>>>(
      xh, Wqh, qh, bq, nullptr, 1,
      Bc * Lc, Cc, Cc, Cc, Cc, Cc, 0, 0, 0, 1.0f,
      0, 0, 0, 0, 0, 0, 1);

  // 3) fused offset head
  compute_weff_kernel<<<1, Cg, 0, stream>>>(W1, b1, W2, b2, weff);
  compute_off_kernel<<<dim3(Lc / 256, BG), 256, 0, stream>>>(qh, weff, b2, offb);

  // 4) deformable bilinear gather -> xs [B,L,C] f16
  gather_xs_kernel<<<dim3(Lc, BG), Cg, 0, stream>>>(x, offb, xsh);

  // 5) K = xs @ Wk^T + bk ; V = xs @ Wv^T + bv + rel_bias
  wmma_gemm_kernel<<<gBig, 128, 0, stream>>>(
      xsh, Wkh, khh, bk, nullptr, 1,
      Bc * Lc, Cc, Cc, Cc, Cc, Cc, 0, 0, 0, 1.0f,
      0, 0, 0, 0, 0, 0, 1);
  wmma_gemm_kernel<<<gBig, 128, 0, stream>>>(
      xsh, Wvh, vhh, bv, rel, Lc,
      Bc * Lc, Cc, Cc, Cc, Cc, Cc, 0, 0, 0, 1.0f,
      0, 0, 0, 0, 0, 0, 1);

  // 6) logits[bh] = (Qh^T x Kh) * C^-0.5 : M=N=64, K=L (transA, transB), z over 64 heads
  wmma_gemm_kernel<<<dim3(1, 1, BH), 128, 0, stream>>>(
      qh, khh, lgts, nullptr, nullptr, 1,
      Dh, Dh, Lc, Cc, Cc, Dh, 1, 1, 1, 1.0f / 32.0f,
      (long long)Lc * Cc, (long long)Dh,
      (long long)Lc * Cc, (long long)Dh,
      (long long)Hc * Dh * Dh, (long long)Dh * Dh, Hc);

  // 7) softmax over channel dim (64)
  softmax64_kernel<<<BH * Dh, 64, 0, stream>>>(lgts, attn);

  // 8) o_t[l,i] = sum_j vh_t[l,j] * attn[i,j] : M=L, N=64, K=64 per head
  wmma_gemm_kernel<<<dim3(Lc / 64, 1, BH), 128, 0, stream>>>(
      vhh, attn, oh, nullptr, nullptr, 1,
      Lc, Dh, Dh, Cc, Dh, Cc, 0, 0, 0, 1.0f,
      (long long)Lc * Cc, (long long)Dh,
      (long long)Hc * Dh * Dh, (long long)Dh * Dh,
      (long long)Lc * Cc, (long long)Dh, Hc);

  // 9) out = o_t @ Wo^T + bo  (f32 to d_out)
  wmma_gemm_kernel<<<gBig, 128, 0, stream>>>(
      oh, Woh, d_out, bo, nullptr, 1,
      Bc * Lc, Cc, Cc, Cc, Cc, Cc, 0, 0, 1, 1.0f,
      0, 0, 0, 0, 0, 0, 1);

  (void)in_sizes; (void)n_in; (void)out_size; (void)ws_size;
}
